// SparseAttention_29497835389696
// MI455X (gfx1250) — matrix operand order repaired and checked by reading, not compiled, not run on hardware
//
#include <hip/hip_runtime.h>

// ---------------------------------------------------------------------------
// SparseAttention for MI455X (gfx1250, wave32, WMMA bf16)
// B=4, H=4, T1=T2=2048, N_FEAT=256, D_K=64, mask all-ones, SCALE=1
// ---------------------------------------------------------------------------

#define B_SZ   4
#define N_HEAD 4
#define N_FEAT 256
#define D_K    64
#define T_SEQ  2048

typedef __attribute__((ext_vector_type(16))) __bf16        v16bf;
typedef __attribute__((ext_vector_type(8)))  float         v8f;
typedef __attribute__((ext_vector_type(4)))  unsigned int  uv4;

union Frag {
  v16bf          v;
  unsigned short u[16];
  uv4            q[2];
};

__device__ __forceinline__ unsigned short f2bf(float x) {
  unsigned int u = __builtin_bit_cast(unsigned int, x);
  u += 0x7FFFu + ((u >> 16) & 1u);          // round-to-nearest-even
  return (unsigned short)(u >> 16);
}

__device__ __forceinline__ v8f v8zero() {
  v8f z = {0.f, 0.f, 0.f, 0.f, 0.f, 0.f, 0.f, 0.f};
  return z;
}

__device__ __forceinline__ v8f wmma_bf16(const Frag& a, const Frag& b, v8f c) {
  return __builtin_amdgcn_wmma_f32_16x16x32_bf16(
      /*neg_a=*/false, a.v, /*neg_b=*/false, b.v,
      /*c_mod=*/(short)0, c, /*reuse_a=*/false, /*reuse_b=*/false);
}

// ---------------------------------------------------------------------------
// Kernel 0: one-time f32 -> bf16 conversion of the three weight matrices.
//   grid = (64, 3), block = 256; each block converts 1024 elements.
// ---------------------------------------------------------------------------
__global__ __launch_bounds__(256) void k_cvt_w(
    const float* __restrict__ Wq, const float* __restrict__ Wk,
    const float* __restrict__ Wv,
    unsigned short* __restrict__ wqb, unsigned short* __restrict__ wkb,
    unsigned short* __restrict__ wvb)
{
  const int y = blockIdx.y;
  const float* src = (y == 0) ? Wq : ((y == 1) ? Wk : Wv);
  unsigned short* dst = (y == 0) ? wqb : ((y == 1) ? wkb : wvb);
  const int base = blockIdx.x * 1024 + threadIdx.x * 4;
#pragma unroll
  for (int i = 0; i < 4; ++i) dst[base + i] = f2bf(src[base + i]);
}

// ---------------------------------------------------------------------------
// Kernel 1: QKV projection (X @ W^T + b), per-head L2 norm for q,k.
//   grid = (B*T/64, 3), block = 256 (8 waves). Each block: 64 rows x 256 cols.
//   Weights read as pre-converted bf16 (pure load path, no conversion ALU).
//   Outputs bf16: qn,kn as (b,h,t,d); v transposed to (b,h,d,t).
//   Dynamic LDS: Out[64][260] f32 = 66,560 B.
// ---------------------------------------------------------------------------
__global__ __launch_bounds__(256) void k_proj(
    const float* __restrict__ qin, const float* __restrict__ kin,
    const float* __restrict__ vin,
    const unsigned short* __restrict__ wqb, const float* __restrict__ bq,
    const unsigned short* __restrict__ wkb, const float* __restrict__ bk,
    const unsigned short* __restrict__ wvb, const float* __restrict__ bv,
    unsigned short* __restrict__ qn, unsigned short* __restrict__ kn,
    unsigned short* __restrict__ vT)
{
  extern __shared__ char smem[];
  float* Out = (float*)smem;                  // [64][260] padded

  const int y   = blockIdx.y;                 // 0=q 1=k 2=v
  const int tid = threadIdx.x;
  const int lane = tid & 31;
  const int w    = tid >> 5;
  const int rowBase = blockIdx.x * 64;

  const float*          X  = (y == 0) ? qin : ((y == 1) ? kin : vin);
  const unsigned short* Wb = (y == 0) ? wqb : ((y == 1) ? wkb : wvb);
  const float*          bs = (y == 0) ? bq  : ((y == 1) ? bk  : bv);

  const int mTile = w >> 1;                   // 0..3
  const int nHalf = w & 1;                    // 0..1  (8 n-tiles each)
  const int nlane = lane & 15;
  const int kh    = lane >> 4;
  const int mRow  = mTile * 16 + nlane;

  v8f acc[8];
#pragma unroll
  for (int nn = 0; nn < 8; ++nn) acc[nn] = v8zero();

  const float* xrow = X + (size_t)(rowBase + mRow) * N_FEAT;

  for (int kc = 0; kc < 8; ++kc) {
    const int kb = kc * 32;
    Frag a;                                    // converted once per 8 WMMAs
    const float* r0 = xrow + kb + kh * 8;
    const float* r1 = xrow + kb + 16 + kh * 8;
#pragma unroll
    for (int j = 0; j < 8; ++j) { a.u[j] = f2bf(r0[j]); a.u[8 + j] = f2bf(r1[j]); }
#pragma unroll
    for (int nn = 0; nn < 8; ++nn) {
      const int o = (nHalf * 8 + nn) * 16 + nlane;        // output feature
      const unsigned short* wp = Wb + (size_t)o * N_FEAT + kb + kh * 8;
      Frag bfr;
      bfr.q[0] = *(const uv4*)(wp);
      bfr.q[1] = *(const uv4*)(wp + 16);
      acc[nn] = wmma_bf16(a, bfr, acc[nn]);
    }
  }

  // store + bias into LDS Out
#pragma unroll
  for (int nn = 0; nn < 8; ++nn) {
    const int col = (nHalf * 8 + nn) * 16 + nlane;
    const float bb = bs[col];
#pragma unroll
    for (int vv = 0; vv < 8; ++vv) {
      const int m = mTile * 16 + vv + 8 * kh;
      Out[m * 260 + col] = acc[nn][vv] + bb;
    }
  }
  __syncthreads();

  const int b     = rowBase >> 11;            // T=2048
  const int tBase = rowBase & (T_SEQ - 1);

  if (y < 2) {
    unsigned short* dst = (y == 0) ? qn : kn;
    const int r = tid >> 2, seg = tid & 3;    // 64 rows x 4 heads
    const float* rowp = Out + r * 260 + seg * 64;
    float ss = 0.f;
#pragma unroll 8
    for (int d = 0; d < 64; ++d) ss += rowp[d] * rowp[d];
    const float rn = rsqrtf(ss);              // SCALE == 1
    unsigned short* dp =
        dst + (((size_t)(b * N_HEAD + seg)) * T_SEQ + (tBase + r)) * D_K;
#pragma unroll 8
    for (int d = 0; d < 64; ++d) dp[d] = f2bf(rowp[d] * rn);
  } else {
    const int h = tid >> 6, d = tid & 63;     // 256 columns
    unsigned short* dp =
        vT + (((size_t)(b * N_HEAD + h)) * D_K + d) * T_SEQ + tBase;
#pragma unroll 8
    for (int r = 0; r < 64; ++r) dp[r] = f2bf(Out[r * 260 + h * 64 + d]);
  }
}

// ---------------------------------------------------------------------------
// Kernel 2: scores (WMMA) -> sparsemax (Michelot, LDS-resident) -> attn@v (WMMA)
//   grid = (T/16, B*H), block = 256 (8 waves). 16 query rows per block.
//   Dynamic LDS: S[16][2064] f32 (132,096) + P[16][2064] bf16 (66,048)
//                + mask[2048] i32 (8,192) + xred[1024] f32 (4,096)
//                = 210,432 B  (<320KB WGP LDS)
// ---------------------------------------------------------------------------
#define SST 2064   // padded row stride (bank-conflict free, 16B aligned)

__global__ __launch_bounds__(256) void k_attn(
    const unsigned short* __restrict__ qn, const unsigned short* __restrict__ kn,
    const unsigned short* __restrict__ vT, const int* __restrict__ mask,
    float* __restrict__ attnOut, float* __restrict__ xOut)
{
  extern __shared__ char smem[];
  float*          S     = (float*)smem;                              // [16][SST]
  unsigned short* P     = (unsigned short*)(smem + 16 * SST * 4);    // [16][SST]
  int*            maskF = (int*)(smem + 16 * SST * 4 + 16 * SST * 2);
  float*          xred  = (float*)(smem + 16 * SST * 4 + 16 * SST * 2 + T_SEQ * 4);

  const int tid  = threadIdx.x;
  const int lane = tid & 31;
  const int w    = tid >> 5;
  const int bh   = blockIdx.y;                // b*H + h
  const int b    = bh >> 2;
  const int t0   = blockIdx.x * 16;

  for (int i = tid; i < T_SEQ; i += 256) maskF[i] = mask[b * T_SEQ + i];
  __syncthreads();

  const int nlane = lane & 15;
  const int kh    = lane >> 4;

  // ---- A-frags for the 16x64 q tile (loaded once, reused across 2048 cols)
  const unsigned short* qrow =
      qn + (((size_t)bh) * T_SEQ + t0 + nlane) * D_K;
  Frag a0, a1;
  a0.q[0] = *(const uv4*)(qrow + kh * 8);
  a0.q[1] = *(const uv4*)(qrow + 16 + kh * 8);
  a1.q[0] = *(const uv4*)(qrow + 32 + kh * 8);
  a1.q[1] = *(const uv4*)(qrow + 48 + kh * 8);

  // ---- score phase: wave w covers n-tiles [w*16, w*16+16)
  for (int nt = w * 16; nt < w * 16 + 16; ++nt) {
    const int n = nt * 16 + nlane;            // key index
    const unsigned short* kp = kn + (((size_t)bh) * T_SEQ + n) * D_K;
    __builtin_prefetch(kp + 16 * D_K, 0, 0);  // next n-tile (global_prefetch_b8)
    Frag b0, b1;
    b0.q[0] = *(const uv4*)(kp + kh * 8);
    b0.q[1] = *(const uv4*)(kp + 16 + kh * 8);
    b1.q[0] = *(const uv4*)(kp + 32 + kh * 8);
    b1.q[1] = *(const uv4*)(kp + 48 + kh * 8);
    v8f c = v8zero();
    c = wmma_bf16(a0, b0, c);
    c = wmma_bf16(a1, b1, c);
    const int f = maskF[n];
#pragma unroll
    for (int vv = 0; vv < 8; ++vv) {
      const int m = vv + 8 * kh;
      S[m * SST + n] = f ? (c[vv] * 0.125f) : -1.0e30f;   // * 1/sqrt(64); mask
    }
  }
  __syncthreads();

  // ---- sparsemax per row: 16 lanes per row, Michelot fixed-point for tau
  {
    const int r = tid >> 4, j = tid & 15;
    const float* Sr = S + r * SST;
    float mx = -3.0e38f;
    for (int i = j; i < T_SEQ; i += 16) mx = fmaxf(mx, Sr[i]);
    mx = fmaxf(mx, __shfl_xor(mx, 1));
    mx = fmaxf(mx, __shfl_xor(mx, 2));
    mx = fmaxf(mx, __shfl_xor(mx, 4));
    mx = fmaxf(mx, __shfl_xor(mx, 8));
    float tau = mx - 1.0f;                    // tau* in [max-1, max): valid start
#pragma unroll 1
    for (int it = 0; it < 16; ++it) {
      float s = 0.f; int c = 0;
      for (int i = j; i < T_SEQ; i += 16) {
        const float z = Sr[i];
        if (z > tau) { s += z; c += 1; }
      }
      s += __shfl_xor(s, 1); c += __shfl_xor(c, 1);
      s += __shfl_xor(s, 2); c += __shfl_xor(c, 2);
      s += __shfl_xor(s, 4); c += __shfl_xor(c, 4);
      s += __shfl_xor(s, 8); c += __shfl_xor(c, 8);
      tau = (s - 1.0f) / (float)c;            // c >= 1 always (tau < max)
    }
    float* Aout = attnOut + (((size_t)bh) * T_SEQ + (t0 + r)) * T_SEQ;
    const float* Srr = S + r * SST;
    unsigned short* Pr = P + r * SST;
    for (int i = j; i < T_SEQ; i += 16) {
      float p = Srr[i] - tau;
      p = p > 0.f ? p : 0.f;                  // masked cols -> 0 automatically
      Pr[i] = f2bf(p);                        // bf16 probs for attn@v
      Aout[i] = p;                            // attn output
    }
  }
  __syncthreads();

  // ---- attn@v: X[16][64] = P[16x2048] @ V[2048x64]; K split across waves
  {
    const int nt  = w & 3;                    // dk tile (4 x 16 cols)
    const int khv = w >> 2;                   // K half (0: 0..1023, 1: 1024..2047)
    const int n   = nt * 16 + nlane;          // dk column
    const unsigned short* vp = vT + (((size_t)bh) * D_K + n) * T_SEQ;
    const unsigned short* Pr = P + nlane * SST;   // A row m = nlane

    v8f c = v8zero();
    for (int kc = 0; kc < 32; ++kc) {
      const int kb = khv * 1024 + kc * 32;
      Frag a;                                  // pure ds_load_b128 path
      a.q[0] = *(const uv4*)(Pr + kb + kh * 8);
      a.q[1] = *(const uv4*)(Pr + kb + 16 + kh * 8);
      Frag bfr;
      const unsigned short* bp = vp + kb + kh * 8;
      bfr.q[0] = *(const uv4*)(bp);
      bfr.q[1] = *(const uv4*)(bp + 16);
      c = wmma_bf16(a, bfr, c);
    }

    if (khv == 1) {
#pragma unroll
      for (int vv = 0; vv < 8; ++vv) xred[nt * 256 + lane * 8 + vv] = c[vv];
    }
    __syncthreads();
    if (khv == 0) {
      const int h = bh & 3;
#pragma unroll
      for (int vv = 0; vv < 8; ++vv) {
        const float val = c[vv] + xred[nt * 256 + lane * 8 + vv];
        const int m = vv + 8 * kh;
        xOut[((size_t)(b * T_SEQ) + t0 + m) * N_FEAT + h * 64 + n] = val;
      }
    }
  }
}

// ---------------------------------------------------------------------------
extern "C" void kernel_launch(void* const* d_in, const int* in_sizes, int n_in,
                              void* d_out, int out_size, void* d_ws, size_t ws_size,
                              hipStream_t stream) {
  (void)in_sizes; (void)n_in; (void)out_size; (void)ws_size;

  const float* query = (const float*)d_in[0];
  const float* key   = (const float*)d_in[1];
  const float* value = (const float*)d_in[2];
  const int*   mask  = (const int*)  d_in[3];
  const float* Wq    = (const float*)d_in[4];
  const float* bq    = (const float*)d_in[5];
  const float* Wk    = (const float*)d_in[6];
  const float* bk    = (const float*)d_in[7];
  const float* Wv    = (const float*)d_in[8];
  const float* bv    = (const float*)d_in[9];

  const size_t perBuf = (size_t)B_SZ * N_HEAD * T_SEQ * D_K;   // 2,097,152 bf16
  const size_t wBuf   = (size_t)N_FEAT * N_FEAT;               // 65,536 bf16
  unsigned short* qn  = (unsigned short*)d_ws;
  unsigned short* kn  = qn + perBuf;
  unsigned short* vT  = kn + perBuf;
  unsigned short* wqb = vT + perBuf;
  unsigned short* wkb = wqb + wBuf;
  unsigned short* wvb = wkb + wBuf;

  float* attnOut = (float*)d_out;                              // (B,H,T,T)
  float* xOut    = attnOut + (size_t)B_SZ * N_HEAD * T_SEQ * T_SEQ;

  // one-time weight conversion to bf16
  k_cvt_w<<<dim3(64, 3), 256, 0, stream>>>(Wq, Wk, Wv, wqb, wkb, wvb);

  // QKV projection + normalize: 67 KB dynamic LDS (>64KB static cap, <320KB WGP)
  k_proj<<<dim3((B_SZ * T_SEQ) / 64, 3), 256, 64 * 260 * 4, stream>>>(
      query, key, value, wqb, bq, wkb, bk, wvb, bv, qn, kn, vT);

  // scores -> sparsemax -> attn@v: 210,432 B dynamic LDS
  const size_t smem2 = 16 * SST * 4 + 16 * SST * 2 + T_SEQ * 4 + 4 * 256 * 4;
  k_attn<<<dim3(T_SEQ / 16, B_SZ * N_HEAD), 256, smem2, stream>>>(
      qn, kn, vT, mask, attnOut, xOut);
}
